// TreeProjector_5720896438577
// MI455X (gfx1250) — compile-verified, matrix operand order checked by reading, not run on hardware
//
#include <hip/hip_runtime.h>

typedef unsigned short u16;
typedef __bf16 bf16_t;
typedef bf16_t v16bf __attribute__((ext_vector_type(16)));
typedef float  v8f   __attribute__((ext_vector_type(8)));

union Frag { uint4 q[2]; v16bf v; };

__device__ __forceinline__ u16 f2bf(float f) {
    unsigned int u = __float_as_uint(f);
    u = u + 0x7FFFu + ((u >> 16) & 1u);     // round to nearest even
    return (u16)(u >> 16);
}

// ---- gfx1250 async global->LDS staging (guarded; falls back to VGPR staging)
#if defined(__has_builtin)
#if __has_builtin(__builtin_amdgcn_global_load_async_to_lds_b128)
#define USE_ASYNC 1
#endif
#endif
#ifndef USE_ASYNC
#define USE_ASYNC 0
#endif

#if USE_ASYNC
// builtin signature (from clang diagnostic): (v4i32 AS1*, v4i32 AS3*, imm, imm)
typedef int gi4 __attribute__((__vector_size__(16)));
typedef __attribute__((address_space(1))) gi4 as1_gi4;
typedef __attribute__((address_space(3))) gi4 as3_gi4;
__device__ __forceinline__ void async_b128(const void* g, void* l) {
    __builtin_amdgcn_global_load_async_to_lds_b128((as1_gi4*)g, (as3_gi4*)l, 0, 0);
}
__device__ __forceinline__ void wait_async0() {
#if __has_builtin(__builtin_amdgcn_s_wait_asynccnt)
    __builtin_amdgcn_s_wait_asynccnt(0);
#else
    asm volatile("s_wait_asynccnt 0" ::: "memory");
#endif
}
#endif

// ---------------------------------------------------------------- elementwise
__global__ void k_zero_f32(float* p, long n) {
    long i = (long)blockIdx.x * blockDim.x + threadIdx.x;
    long s = (long)gridDim.x * blockDim.x;
    for (; i < n; i += s) p[i] = 0.f;
}

// f32 [n,C] -> bf16 [n,ld] (pad columns untouched; zero them with k_zero_pad)
__global__ void k_f32_to_bf16_pad(const float* __restrict__ s, int n, int C,
                                  u16* __restrict__ d, int ld) {
    long i = (long)blockIdx.x * blockDim.x + threadIdx.x;
    if (i >= (long)n * C) return;
    int r = (int)(i / C), c = (int)(i % C);
    d[(size_t)r * ld + c] = f2bf(s[i]);
}

__global__ void k_zero_pad(u16* __restrict__ buf, int n, int c0, int ld) {
    int w = ld - c0;
    long i = (long)blockIdx.x * blockDim.x + threadIdx.x;
    if (i >= (long)n * w) return;
    int r = (int)(i / w), c = (int)(i % w);
    buf[(size_t)r * ld + c0 + c] = 0;
}

__global__ void k_copy_cols(const u16* __restrict__ s, int n, int C, int lds_,
                            u16* __restrict__ d, int ldo, int coff) {
    long i = (long)blockIdx.x * blockDim.x + threadIdx.x;
    if (i >= (long)n * C) return;
    int r = (int)(i / C), c = (int)(i % C);
    d[(size_t)r * ldo + coff + c] = s[(size_t)r * lds_ + c];
}

// weights f32 [K][Cin][Cout] -> bf16 transposed [K][Cout][ldk], zero K-tail
__global__ void k_wt_bf16(const float* __restrict__ w, u16* __restrict__ wt,
                          int K, int Cin, int Cout, int ldk) {
    long i = (long)blockIdx.x * blockDim.x + threadIdx.x;
    long tot = (long)K * Cout * ldk;
    if (i >= tot) return;
    int kk = (int)(i % ldk);
    long rem = i / ldk;
    int co = (int)(rem % Cout);
    int k  = (int)(rem / Cout);
    u16 v = 0;
    if (kk < Cin) v = f2bf(w[((size_t)k * Cin + kk) * Cout + co]);
    wt[i] = v;
}

// ---------------------------------------------------------------- batchnorm
__global__ void k_bn_partial(const float* __restrict__ acc, int n, int C,
                             float* __restrict__ sum, float* __restrict__ sumsq) {
    int c = blockIdx.x * blockDim.x + threadIdx.x;
    if (c >= C) return;
    int nch = gridDim.y;
    int per = (n + nch - 1) / nch;
    int r0 = blockIdx.y * per;
    int r1 = r0 + per; if (r1 > n) r1 = n;
    float s = 0.f, s2 = 0.f;
    for (int r = r0; r < r1; ++r) {
        float v = acc[(size_t)r * C + c];
        s += v; s2 += v * v;
    }
    atomicAdd(&sum[c], s);
    atomicAdd(&sumsq[c], s2);
}

__global__ void k_bn_finalize(const float* __restrict__ sum, const float* __restrict__ sumsq,
                              int n, int C, const float* __restrict__ g,
                              const float* __restrict__ b,
                              float* __restrict__ scale, float* __restrict__ shift) {
    int c = blockIdx.x * blockDim.x + threadIdx.x;
    if (c >= C) return;
    float inv = 1.f / (float)n;
    float mu  = sum[c] * inv;
    float var = sumsq[c] * inv - mu * mu;
    float sc  = g[c] * rsqrtf(var + 1e-5f);
    scale[c] = sc;
    shift[c] = b[c] - mu * sc;
}

__global__ void k_bn_apply(const float* __restrict__ acc, int n, int C,
                           const float* __restrict__ scale, const float* __restrict__ shift,
                           u16* __restrict__ out, int ldo, int coff) {
    long i = (long)blockIdx.x * blockDim.x + threadIdx.x;
    if (i >= (long)n * C) return;
    int r = (int)(i / C), c = (int)(i % C);
    float v = acc[i] * scale[c] + shift[c];
    v = v > 0.f ? v : 0.f;
    out[(size_t)r * ldo + coff + c] = f2bf(v);
}

// ---------------------------------------------------------------- spconv WMMA
// Rulebook sparse conv, one kernel-offset per blockIdx.z:
//   acc[oi[k][p], :] += bf16(X[ii[k][p], :]) @ W[k]      (f32 accumulate)
// X: bf16, row stride ldx (multiple of 32, K-tail zeroed)
// Wt: bf16, [K][Cout][ldx] (pre-transposed, K-tail zeroed)
// Tile 256x64 per 256-thread block; wave w owns M-subtiles w and w+8, sweeps
// 4 N-subtiles => 8 v_wmma_f32_16x16x32_bf16 per K-step sharing B fragments.
// Double-buffered LDS; staging via async global->LDS b128 when available
// (ASYNCcnt pipeline), else b128 VGPR staging. OOB gather rows / B columns
// clamp to row 0 — their outputs are dropped by the scatter guard.
#define BM  256
#define LDP 40

__global__ __launch_bounds__(256)
void k_spconv_wmma(const u16* __restrict__ X, int ldx,
                   const u16* __restrict__ Wt,
                   const int* __restrict__ ii, const int* __restrict__ oi,
                   int P, int Cout, float* __restrict__ acc) {
    __shared__ u16 lsA[2][BM * LDP];
    __shared__ u16 lsB[2][64 * LDP];

    const int tid  = threadIdx.x;
    const int lane = tid & 31;
    const int wv   = tid >> 5;
    const int p0   = blockIdx.x * BM;
    const int n0   = blockIdx.y * 64;
    const size_t kz = blockIdx.z;

    const int* iik = ii ? ii + kz * (size_t)P : nullptr;
    const int* oik = oi ? oi + kz * (size_t)P : nullptr;

    // A loader: thread tid stages gathered row tid (32 elems = 4 x b128)
    int agr = 0;
    if (p0 + tid < P) agr = iik ? iik[p0 + tid] : (p0 + tid);
    const u16* Arow = X + (size_t)agr * ldx;

    // B loader: thread -> (col bn_, 8-elem K run); Wt rows contiguous in K
    const int bn_  = tid >> 2;
    const int bk0  = (tid & 3) * 8;
    const int bcol = (n0 + bn_ < Cout) ? (n0 + bn_) : 0;
    const u16* Brow = Wt + ((size_t)kz * Cout + bcol) * ldx;

    v8f c0[4] = {}, c1[4] = {};
    const int mrow = wv * 16 + (lane & 15);
    const int eb   = (lane < 16) ? 0 : 8;   // K-base of this lane's two runs
    const int pn   = lane & 15;
    const int nsteps = ldx >> 5;

    auto compute = [&](int cur) {
        Frag a0, a1;
        const u16* A0 = &lsA[cur][mrow * LDP + eb];
        const u16* A1 = &lsA[cur][(mrow + 128) * LDP + eb];
        a0.q[0] = *(const uint4*)(A0);
        a0.q[1] = *(const uint4*)(A0 + 16);
        a1.q[0] = *(const uint4*)(A1);
        a1.q[1] = *(const uint4*)(A1 + 16);
#pragma unroll
        for (int nt = 0; nt < 4; ++nt) {
            const u16* Bb = &lsB[cur][(nt * 16 + pn) * LDP + eb];
            Frag b;
            b.q[0] = *(const uint4*)(Bb);
            b.q[1] = *(const uint4*)(Bb + 16);
            c0[nt] = __builtin_amdgcn_wmma_f32_16x16x32_bf16(
                false, a0.v, false, b.v, (short)0, c0[nt], false, false);
            c1[nt] = __builtin_amdgcn_wmma_f32_16x16x32_bf16(
                false, a1.v, false, b.v, (short)0, c1[nt], false, false);
        }
    };

#if USE_ASYNC
    auto issue = [&](int buf, int k0) {
        u16* la = &lsA[buf][tid * LDP];
        const u16* ga = Arow + k0;
#pragma unroll
        for (int j = 0; j < 4; ++j) async_b128(ga + 8 * j, la + 8 * j);
        async_b128(Brow + k0 + bk0, &lsB[buf][bn_ * LDP + bk0]);
    };
    issue(0, 0);
    for (int t = 0; t < nsteps; ++t) {
        const int cur = t & 1;
        wait_async0();          // own wave's tile-t transfers complete
        __syncthreads();        // everyone's transfers visible
        if (t + 1 < nsteps) issue(cur ^ 1, (t + 1) << 5);
        compute(cur);
    }
#else
    uint4 ra[4], rb;
    auto fetch = [&](int k0) {
        const uint4* pa = (const uint4*)(Arow + k0);
#pragma unroll
        for (int j = 0; j < 4; ++j) ra[j] = pa[j];
        rb = *(const uint4*)(Brow + k0 + bk0);
    };
    auto store = [&](int buf) {
#pragma unroll
        for (int j = 0; j < 4; ++j)
            *(uint4*)&lsA[buf][tid * LDP + 8 * j] = ra[j];
        *(uint4*)&lsB[buf][bn_ * LDP + bk0] = rb;
    };
    fetch(0);
    store(0);
    for (int t = 0; t < nsteps; ++t) {
        const int cur = t & 1;
        const bool more = (t + 1) < nsteps;
        if (more) fetch((t + 1) << 5);
        __syncthreads();
        compute(cur);
        if (more) store(cur ^ 1);
    }
#endif

    // Scatter-add through rulebook (C/D layout: VGPR r -> M=r(+8), N=lane&15)
    const int mb = (lane < 16) ? 0 : 8;
#pragma unroll
    for (int g = 0; g < 2; ++g) {
        const v8f* cc = g ? c1 : c0;
        const int pbase = p0 + g * 128 + wv * 16 + mb;
#pragma unroll
        for (int nt = 0; nt < 4; ++nt) {
#pragma unroll
            for (int r = 0; r < 8; ++r) {
                int p = pbase + r;
                int n = n0 + nt * 16 + pn;
                if (p < P && n < Cout) {
                    int orow = oik ? oik[p] : p;
                    atomicAdd(&acc[(size_t)orow * Cout + n], cc[nt][r]);
                }
            }
        }
    }
}

// ---------------------------------------------------------------- host driver
extern "C" void kernel_launch(void* const* d_in, const int* in_sizes, int n_in,
                              void* d_out, int out_size, void* d_ws, size_t ws_size,
                              hipStream_t stream) {
    (void)in_sizes; (void)n_in; (void)out_size; (void)ws_size;

    static const int NL[5] = {50000, 12500, 3200, 800, 200};
    static const int CHN[4] = {64, 128, 256, 512};

    const float* feats = (const float*)d_in[0];
    auto ENC = [&](int i, int j) { return d_in[1 + 9 * i + j]; };   // w1,g1,b1,w2,g2,b2,w3,g3,b3
    auto DEC = [&](int i, int j) { return d_in[37 + 3 * i + j]; };  // w,g,b
    const void* smooth_w = d_in[49];
    const void* smooth_g = d_in[50];
    const void* smooth_b = d_in[51];
    const void* sem_w = d_in[52];
    const void* ins_w = d_in[53];
    auto RUL = [&](int i, int j) { return (const int*)d_in[54 + 6 * i + j]; }; // ci,co,di,do,ui,uo

    // deterministic bump allocator over d_ws
    char* ws = (char*)d_ws;
    size_t off = 0;
    auto alloc = [&](size_t bytes) -> void* {
        void* p = ws + off;
        off = (off + bytes + 255) & ~(size_t)255;
        return p;
    };

    u16* d0 = (u16*)alloc((size_t)50000 * 32 * 2);          // feats, ld=32 (pad of 4)
    u16* dskip[5]; dskip[0] = d0;
    for (int i = 1; i <= 4; ++i) dskip[i] = (u16*)alloc((size_t)NL[i] * CHN[i - 1] * 2);
    u16* xa = (u16*)alloc((size_t)50000 * 64 * 2);
    u16* xb = (u16*)alloc((size_t)50000 * 64 * 2);
    const int RW[4] = {964, 960, 896, 768};                 // real concat widths
    const int PW[4] = {992, 960, 896, 768};                 // padded strides
    u16* rbuf[4];
    for (int i = 0; i < 4; ++i) rbuf[i] = (u16*)alloc((size_t)NL[i] * PW[i] * 2);
    u16* lat  = (u16*)alloc((size_t)50000 * 512 * 2);
    float* acc = (float*)alloc((size_t)50000 * 960 * 4);    // reused accumulator
    u16* bw   = (u16*)alloc((size_t)27 * 512 * 992 * 2);    // reused transposed bf16 weights
    float* ss = (float*)alloc(4096 * 4);                    // sum|sumsq|scale|shift

    auto blks = [](long n) { return dim3((unsigned)((n + 255) / 256)); };
    auto zero = [&](float* p, size_t n) {
        k_zero_f32<<<dim3(2048), 256, 0, stream>>>(p, (long)n);
    };
    auto spconv = [&](const u16* X, int Cin, int ldx, const void* Wf32, int K,
                      const int* ik, const int* ok, int P, int Cout, float* ac) {
        long wtot = (long)K * Cout * ldx;
        k_wt_bf16<<<blks(wtot), 256, 0, stream>>>((const float*)Wf32, bw, K, Cin, Cout, ldx);
        zero(ac, (size_t)P * Cout);
        dim3 g((unsigned)((P + BM - 1) / BM), (unsigned)((Cout + 63) / 64), (unsigned)K);
        k_spconv_wmma<<<g, 256, 0, stream>>>(X, ldx, bw, ik, ok, P, Cout, ac);
    };
    auto bnrelu = [&](float* ac, int n, int C, const void* g, const void* b,
                      u16* outp, int ldo, int coff) {
        zero(ss, 2048);
        dim3 gp((unsigned)((C + 255) / 256), 64);
        k_bn_partial<<<gp, 256, 0, stream>>>(ac, n, C, ss, ss + 1024);
        k_bn_finalize<<<dim3((unsigned)((C + 255) / 256)), 256, 0, stream>>>(
            ss, ss + 1024, n, C, (const float*)g, (const float*)b, ss + 2048, ss + 3072);
        k_bn_apply<<<blks((long)n * C), 256, 0, stream>>>(
            ac, n, C, ss + 2048, ss + 3072, outp, ldo, coff);
    };
    auto copycols = [&](const u16* s, int n, int C, int lds_, u16* d, int ldo, int coff) {
        k_copy_cols<<<blks((long)n * C), 256, 0, stream>>>(s, n, C, lds_, d, ldo, coff);
    };

    // ---- featurize: f32 [50000,4] -> bf16 [50000,32], pad zeroed ----
    k_f32_to_bf16_pad<<<blks((long)50000 * 4), 256, 0, stream>>>(feats, 50000, 4, d0, 32);
    k_zero_pad<<<blks((long)50000 * 28), 256, 0, stream>>>(d0, 50000, 4, 32);

    // ---- encoder ----
    const u16* xin = d0;
    int Cin = 4, ldin = 32;
    for (int l = 0; l < 4; ++l) {
        int C = CHN[l], P = NL[l];
        spconv(xin, Cin, ldin, ENC(l, 0), 27, RUL(l, 0), RUL(l, 1), P, C, acc);
        bnrelu(acc, P, C, ENC(l, 1), ENC(l, 2), xa, C, 0);
        spconv(xa, C, C, ENC(l, 3), 27, RUL(l, 0), RUL(l, 1), P, C, acc);
        bnrelu(acc, P, C, ENC(l, 4), ENC(l, 5), xb, C, 0);
        spconv(xb, C, C, ENC(l, 6), 8, RUL(l, 2), RUL(l, 3), NL[l + 1], C, acc);
        bnrelu(acc, NL[l + 1], C, ENC(l, 7), ENC(l, 8), dskip[l + 1], C, 0);
        xin = dskip[l + 1];
        Cin = C; ldin = C;
    }

    // ---- decoder (transposed conv + BN/ReLU, then skip concat) ----
    const u16* res = dskip[4];
    int Cres = 512, ldres = 512;
    const int SC[4]  = {4, 64, 128, 256};   // skip channels at level i
    const int SLD[4] = {32, 64, 128, 256};  // skip source strides
    for (int i = 3; i >= 0; --i) {
        int s = Cres;  // dec conv is s -> s
        spconv(res, Cres, ldres, DEC(i, 0), 8, RUL(i, 4), RUL(i, 5), NL[i], s, acc);
        bnrelu(acc, NL[i], s, DEC(i, 1), DEC(i, 2), rbuf[i], PW[i], 0);
        copycols(dskip[i], NL[i], SC[i], SLD[i], rbuf[i], PW[i], s);
        if (PW[i] > RW[i])
            k_zero_pad<<<blks((long)NL[i] * (PW[i] - RW[i])), 256, 0, stream>>>(
                rbuf[i], NL[i], RW[i], PW[i]);
        res = rbuf[i];
        Cres = RW[i]; ldres = PW[i];
    }

    // ---- smooth conv 964 -> 512 ----
    spconv(rbuf[0], 964, 992, smooth_w, 27, RUL(0, 0), RUL(0, 1), 50000, 512, acc);
    bnrelu(acc, 50000, 512, smooth_g, smooth_b, lat, 512, 0);

    // ---- heads: dense 1x1 matmuls via identity-rule spconv, f32 out ----
    float* out = (float*)d_out;
    spconv(lat, 512, 512, sem_w, 1, nullptr, nullptr, 50000, 4, out);
    spconv(lat, 512, 512, ins_w, 1, nullptr, nullptr, 50000, 128, out + (size_t)50000 * 4);
}
